// MiniGPT_30932354466023
// MI455X (gfx1250) — matrix operand order repaired and checked by reading, not compiled, not run on hardware
//
#include <hip/hip_runtime.h>

#define SEQ      2048
#define HIDDEN   1024
#define VOCAB    50259
#define NUM_HEADS 16
#define HEAD_DIM  64

typedef __bf16 bf16_t;
typedef __attribute__((ext_vector_type(16))) __bf16 v16bf;
typedef __attribute__((ext_vector_type(8)))  float  v8f;

static __device__ __forceinline__ v8f wmma_bf16(v16bf a, v16bf b, v8f c) {
  // D = A(16x32 bf16) x B(32x16 bf16) + C(16x16 f32)
  return __builtin_amdgcn_wmma_f32_16x16x32_bf16(false, a, false, b, (short)0, c,
                                                 false, false);
}

// Build a v16bf A/B operand from two 16-byte chunks.
static __device__ __forceinline__ v16bf load_op16(const bf16_t* p0, const bf16_t* p1) {
  v16bf v;
  ((uint4*)&v)[0] = *(const uint4*)p0;
  ((uint4*)&v)[1] = *(const uint4*)p1;
  return v;
}

// ---------------- elementwise: f32 -> bf16 (4 elems / thread) ----------------
__global__ __launch_bounds__(256)
void cvt_f32_bf16(const float* __restrict__ in, bf16_t* __restrict__ out, long n4) {
  long i = (long)blockIdx.x * blockDim.x + threadIdx.x;
  if (i >= n4) return;
  float4 f = ((const float4*)in)[i];
  __align__(8) bf16_t o[4] = {(bf16_t)f.x, (bf16_t)f.y, (bf16_t)f.z, (bf16_t)f.w};
  *(uint2*)(out + i * 4) = *(uint2*)o;
}

// ---------------- embedding gather: emb[s,:] = tok_emb[ids[s],:] ----------------
__global__ __launch_bounds__(256)
void embed_kernel(const int* __restrict__ ids, const float* __restrict__ tok,
                  float* __restrict__ emb) {
  int s  = blockIdx.x;
  int id = ids[s];
  const float4* src = (const float4*)(tok + (long)id * HIDDEN);
  float4*       dst = (float4*)(emb + (long)s * HIDDEN);
  for (int i = threadIdx.x; i < HIDDEN / 4; i += blockDim.x) dst[i] = src[i];
}

// ---------------- prep: xb = bf16(emb); xpe = bf16(emb + pos) (optional) ----------------
__global__ __launch_bounds__(256)
void prep_kernel(const float* __restrict__ emb, const float* __restrict__ pos,
                 bf16_t* __restrict__ xb, bf16_t* __restrict__ xpe) {
  long i = (long)blockIdx.x * blockDim.x + threadIdx.x;  // index of float4
  float4 e = ((const float4*)emb)[i];
  __align__(8) bf16_t o[4] = {(bf16_t)e.x, (bf16_t)e.y, (bf16_t)e.z, (bf16_t)e.w};
  *(uint2*)(xb + i * 4) = *(uint2*)o;
  if (xpe) {
    float4 p = ((const float4*)pos)[i];  // pos table is (SEQ, HIDDEN), same flat layout
    __align__(8) bf16_t q[4] = {(bf16_t)(e.x + p.x), (bf16_t)(e.y + p.y),
                                (bf16_t)(e.z + p.z), (bf16_t)(e.w + p.w)};
    *(uint2*)(xpe + i * 4) = *(uint2*)q;
  }
}

// ---------------- generic WMMA GEMM: C = A(MxK) x B(NxK)^T [+ residual] ----------------
// A, B row-major bf16 with leading dim K. 128x128x32 tiles, 8 waves,
// double-buffered LDS, one barrier per K-step, register-staged global prefetch.
#define BM 128
#define BN 128
#define BK 32
#define LDS_S 40  // element stride (80 B): pads banks, keeps 16-B alignment

__global__ __launch_bounds__(256)
void gemm_bf16(const bf16_t* __restrict__ A, const bf16_t* __restrict__ B,
               int M, int N, int K,
               float* __restrict__ Cf, bf16_t* __restrict__ Cb,
               const float* __restrict__ Res) {
  __shared__ __align__(16) bf16_t As[2][BM * LDS_S];
  __shared__ __align__(16) bf16_t Bs[2][BN * LDS_S];

  const int tid  = threadIdx.x;
  const int lane = tid & 31;
  const int w    = tid >> 5;
  const int wm   = (w & 1) * 64;   // 2 waves along M
  const int wn   = (w >> 1) * 32;  // 4 waves along N
  const int m0   = blockIdx.y * BM;
  const int n0   = blockIdx.x * BN;
  const int hi   = lane >> 4;
  const int lo   = lane & 15;
  const int kb   = hi * 8;         // A-operand K-chunk base for this lane half

  // staging coordinates: each thread copies two 16-B chunks for A and two for B
  const int idx0 = tid * 8;          // tile-element index of chunk 0
  const int idx1 = (tid + 256) * 8;  // tile-element index of chunk 1
  const int r0 = idx0 >> 5, c0 = idx0 & 31;
  const int r1 = idx1 >> 5, c1 = idx1 & 31;
  const int nA0 = m0 + r0, nA1 = m0 + r1;
  const int nB0 = (n0 + r0 < N) ? (n0 + r0) : (N - 1);  // branchless clamp
  const int nB1 = (n0 + r1 < N) ? (n0 + r1) : (N - 1);
  const bf16_t* a0 = A + (long)nA0 * K + c0;
  const bf16_t* a1 = A + (long)nA1 * K + c1;
  const bf16_t* b0 = B + (long)nB0 * K + c0;
  const bf16_t* b1 = B + (long)nB1 * K + c1;

  v8f acc[4][2];
#pragma unroll
  for (int i = 0; i < 4; ++i)
#pragma unroll
    for (int j = 0; j < 2; ++j) acc[i][j] = (v8f)0.0f;

  // preload first K-slab into registers
  uint4 ra0 = *(const uint4*)a0, ra1 = *(const uint4*)a1;
  uint4 rb0 = *(const uint4*)b0, rb1 = *(const uint4*)b1;

  const int nIter = K / BK;
  for (int it = 0; it < nIter; ++it) {
    const int cur = it & 1;
    bf16_t* pAs = As[cur];
    bf16_t* pBs = Bs[cur];
    *(uint4*)(&pAs[r0 * LDS_S + c0]) = ra0;
    *(uint4*)(&pAs[r1 * LDS_S + c1]) = ra1;
    *(uint4*)(&pBs[r0 * LDS_S + c0]) = rb0;
    *(uint4*)(&pBs[r1 * LDS_S + c1]) = rb1;

    if (it + 1 < nIter) {  // fetch next slab into registers (overlaps with WMMA below)
      a0 += BK; a1 += BK; b0 += BK; b1 += BK;
      ra0 = *(const uint4*)a0;
      ra1 = *(const uint4*)a1;
      rb0 = *(const uint4*)b0;
      rb1 = *(const uint4*)b1;
      __builtin_prefetch(a0 + BK, 0, 3);  // hint the slab after that toward L2
      __builtin_prefetch(b0 + BK, 0, 3);
    }
    __syncthreads();  // single barrier/iter: separates prev reads from next stores

    v16bf a[4], b[2];
#pragma unroll
    for (int i = 0; i < 4; ++i) {
      const bf16_t* p = &pAs[(wm + i * 16 + lo) * LDS_S];
      a[i] = load_op16(p + kb, p + 16 + kb);
    }
#pragma unroll
    for (int j = 0; j < 2; ++j) {
      const bf16_t* p = &pBs[(wn + j * 16 + lo) * LDS_S];
      b[j] = load_op16(p + kb, p + 16 + kb);
    }
#pragma unroll
    for (int i = 0; i < 4; ++i)
#pragma unroll
      for (int j = 0; j < 2; ++j) acc[i][j] = wmma_bf16(a[i], b[j], acc[i][j]);
  }

  // epilogue: C layout — VGPR r: lanes 0-15 -> M=r, lanes 16-31 -> M=8+r; N=lane&15
#pragma unroll
  for (int i = 0; i < 4; ++i)
#pragma unroll
    for (int j = 0; j < 2; ++j)
#pragma unroll
      for (int r = 0; r < 8; ++r) {
        int row = m0 + wm + i * 16 + r + 8 * hi;
        int col = n0 + wn + j * 16 + lo;
        if (row < M && col < N) {
          float v = acc[i][j][r];
          if (Res) v += Res[(long)row * N + col];
          if (Cf) Cf[(long)row * N + col] = v;
          if (Cb) Cb[(long)row * N + col] = (bf16_t)v;
        }
      }
}

// ---------------- causal flash attention, one wave per (head, 16-query tile) ----------------
// QK: (SEQ, 2*HIDDEN) bf16 — cols [0,1024) = Q, [1024,2048) = K (head h at h*64)
// V:  (SEQ, HIDDEN) bf16 ; Out: (SEQ, HIDDEN) bf16 (heads concatenated)
__global__ __launch_bounds__(32)
void attn_kernel(const bf16_t* __restrict__ QK, const bf16_t* __restrict__ V,
                 bf16_t* __restrict__ Out) {
  __shared__ __align__(16) bf16_t Pld[16 * 32];

  const int qt = blockIdx.x, h = blockIdx.y;
  const int q0   = qt * 16;
  const int lane = threadIdx.x;
  const int hi   = lane >> 4;
  const int lo   = lane & 15;
  const int kb   = hi * 8;
  const int ldqk = 2 * HIDDEN;

  // Q A-operands: 2 chunks over head_dim (K=0..31, 32..63)
  v16bf qa[2];
#pragma unroll
  for (int c = 0; c < 2; ++c) {
    const bf16_t* p = QK + (long)(q0 + lo) * ldqk + h * HEAD_DIM + c * 32;
    qa[c] = load_op16(p + kb, p + 16 + kb);
  }

  v8f O[4];
#pragma unroll
  for (int d = 0; d < 4; ++d) O[d] = (v8f)0.0f;
  float m[8], l[8];
#pragma unroll
  for (int r = 0; r < 8; ++r) { m[r] = -3.0e38f; l[r] = 0.0f; }

  for (int k0 = 0; k0 < q0 + 16; k0 += 32) {  // 32-key chunks
    // S = Q x K^T : two 16-key tiles, accumulate over head_dim
    v8f S[2];
    S[0] = (v8f)0.0f; S[1] = (v8f)0.0f;
#pragma unroll
    for (int t = 0; t < 2; ++t)
#pragma unroll
      for (int c = 0; c < 2; ++c) {
        const bf16_t* p = QK + (long)(k0 + t * 16 + lo) * ldqk + HIDDEN +
                          h * HEAD_DIM + c * 32;
        v16bf bv = load_op16(p + kb, p + 16 + kb);
        S[t] = wmma_bf16(qa[c], bv, S[t]);
      }

    // online softmax (rows live at (vgpr r, lane-half hi); cols = lo [+16])
#pragma unroll
    for (int r = 0; r < 8; ++r) {
      int qrow = q0 + r + 8 * hi;
      float s0 = S[0][r] * 0.125f;  // 1/sqrt(64)
      float s1 = S[1][r] * 0.125f;
      s0 = (k0 + lo      > qrow) ? -10000.0f : s0;  // reference NEG_INF
      s1 = (k0 + 16 + lo > qrow) ? -10000.0f : s1;
      float t = fmaxf(s0, s1);
      t = fmaxf(t, __shfl_xor(t, 1, 32));
      t = fmaxf(t, __shfl_xor(t, 2, 32));
      t = fmaxf(t, __shfl_xor(t, 4, 32));
      t = fmaxf(t, __shfl_xor(t, 8, 32));
      float mn = fmaxf(m[r], t);
      float al = __expf(m[r] - mn);
      float p0 = __expf(s0 - mn);
      float p1 = __expf(s1 - mn);
      float rs = p0 + p1;
      rs += __shfl_xor(rs, 1, 32);
      rs += __shfl_xor(rs, 2, 32);
      rs += __shfl_xor(rs, 4, 32);
      rs += __shfl_xor(rs, 8, 32);
      l[r] = l[r] * al + rs;
      m[r] = mn;
      int row = r + 8 * hi;
      Pld[row * 32 + lo]      = (bf16_t)p0;
      Pld[row * 32 + 16 + lo] = (bf16_t)p1;
#pragma unroll
      for (int d = 0; d < 4; ++d) O[d][r] *= al;
    }
    __syncthreads();

    // P as A-operand (16 q-rows x 32 keys)
    v16bf pa;
    {
      const bf16_t* p = &Pld[lo * 32];
      pa = load_op16(p + kb, p + 16 + kb);
    }
    __syncthreads();

    // O += P x V : 4 N-tiles over head_dim
#pragma unroll
    for (int d = 0; d < 4; ++d) {
      v16bf bv;
      bf16_t* bp = (bf16_t*)&bv;
#pragma unroll
      for (int j = 0; j < 16; ++j)
        bp[j] = V[(long)(k0 + kb + j + (j >= 8 ? 8 : 0)) * HIDDEN + h * HEAD_DIM + d * 16 + lo];
      O[d] = wmma_bf16(pa, bv, O[d]);
    }
  }

  // normalize and store combined (s, h*64 + d)
#pragma unroll
  for (int d = 0; d < 4; ++d)
#pragma unroll
    for (int r = 0; r < 8; ++r) {
      int row = q0 + r + 8 * hi;
      Out[(long)row * HIDDEN + h * HEAD_DIM + d * 16 + lo] = (bf16_t)(O[d][r] / l[r]);
    }
}

// ---------------- host orchestration ----------------
extern "C" void kernel_launch(void* const* d_in, const int* in_sizes, int n_in,
                              void* d_out, int out_size, void* d_ws, size_t ws_size,
                              hipStream_t stream) {
  (void)in_sizes; (void)n_in; (void)out_size; (void)ws_size;
  const int*   ids     = (const int*)d_in[0];
  const float* tok_emb = (const float*)d_in[1];
  const float* pos_emb = (const float*)d_in[2];
  const float* Wqkv[2] = {(const float*)d_in[3], (const float*)d_in[5]};
  const float* Wo[2]   = {(const float*)d_in[4], (const float*)d_in[6]};
  float* out = (float*)d_out;

  char*  ws  = (char*)d_ws;
  size_t off = 0;
  auto alloc = [&](size_t bytes) -> void* {
    void* p = ws + off;
    off += (bytes + 255) & ~(size_t)255;
    return p;
  };
  bf16_t* tokemb_b = (bf16_t*)alloc((size_t)VOCAB * HIDDEN * 2);
  bf16_t* wqkv_b[2], *wo_b[2];
  for (int lyr = 0; lyr < 2; ++lyr) {
    wqkv_b[lyr] = (bf16_t*)alloc((size_t)3 * HIDDEN * HIDDEN * 2);
    wo_b[lyr]   = (bf16_t*)alloc((size_t)HIDDEN * HIDDEN * 2);
  }
  float*  emb  = (float*)alloc((size_t)SEQ * HIDDEN * 4);
  bf16_t* xb   = (bf16_t*)alloc((size_t)SEQ * HIDDEN * 2);
  bf16_t* xpe  = (bf16_t*)alloc((size_t)SEQ * HIDDEN * 2);
  bf16_t* qk   = (bf16_t*)alloc((size_t)SEQ * 2 * HIDDEN * 2);
  bf16_t* vb   = (bf16_t*)alloc((size_t)SEQ * HIDDEN * 2);
  bf16_t* comb = (bf16_t*)alloc((size_t)SEQ * HIDDEN * 2);

  // bf16 copies of all weights (tok_emb bf16 = 103 MB -> L2-resident for logits GEMM)
  {
    long n4 = (long)VOCAB * HIDDEN / 4;
    cvt_f32_bf16<<<(n4 + 255) / 256, 256, 0, stream>>>(tok_emb, tokemb_b, n4);
  }
  for (int lyr = 0; lyr < 2; ++lyr) {
    long n4q = (long)3 * HIDDEN * HIDDEN / 4;
    long n4o = (long)HIDDEN * HIDDEN / 4;
    cvt_f32_bf16<<<(n4q + 255) / 256, 256, 0, stream>>>(Wqkv[lyr], wqkv_b[lyr], n4q);
    cvt_f32_bf16<<<(n4o + 255) / 256, 256, 0, stream>>>(Wo[lyr], wo_b[lyr], n4o);
  }

  // embedding gather
  embed_kernel<<<SEQ, 256, 0, stream>>>(ids, tok_emb, emb);

  const int prep_blocks = (SEQ * HIDDEN / 4) / 256;
  for (int lyr = 0; lyr < 2; ++lyr) {
    prep_kernel<<<prep_blocks, 256, 0, stream>>>(emb, pos_emb, xb, xpe);
    // qk = (x+pe) @ Wqkv[:2048]^T   (M=2048, N=2048, K=1024)
    gemm_bf16<<<dim3(2 * HIDDEN / BN, SEQ / BM), 256, 0, stream>>>(
        xpe, wqkv_b[lyr], SEQ, 2 * HIDDEN, HIDDEN, nullptr, qk, nullptr);
    // v = x @ Wqkv[2048:3072]^T     (N=1024)
    gemm_bf16<<<dim3(HIDDEN / BN, SEQ / BM), 256, 0, stream>>>(
        xb, wqkv_b[lyr] + (size_t)2 * HIDDEN * HIDDEN, SEQ, HIDDEN, HIDDEN,
        nullptr, vb, nullptr);
    // causal attention
    attn_kernel<<<dim3(SEQ / 16, NUM_HEADS), 32, 0, stream>>>(qk, vb, comb);
    // emb = emb + comb @ Wo^T  (in-place residual: each element owned by one thread)
    gemm_bf16<<<dim3(HIDDEN / BN, SEQ / BM), 256, 0, stream>>>(
        comb, wo_b[lyr], SEQ, HIDDEN, HIDDEN, emb, nullptr, emb);
  }

  // logits = emb @ tok_emb^T  (M=2048, N=50259, K=1024) -> d_out (f32)
  prep_kernel<<<prep_blocks, 256, 0, stream>>>(emb, nullptr, xb, nullptr);
  gemm_bf16<<<dim3((VOCAB + BN - 1) / BN, SEQ / BM), 256, 0, stream>>>(
      xb, tokemb_b, SEQ, VOCAB, HIDDEN, out, nullptr, nullptr);
}
